// HeterogeneousGINRegressor_81552839016474
// MI455X (gfx1250) — matrix operand order repaired and checked by reading, not compiled, not run on hardware
//
#include <hip/hip_runtime.h>
#include <hip/hip_bf16.h>
#include <stddef.h>

typedef __bf16 bf16_t;
typedef bf16_t v16bf __attribute__((ext_vector_type(16)));
typedef float  v8f   __attribute__((ext_vector_type(8)));

#define NNODES 20000
#define FIN    64
#define HDIM   512
#define NTYPES 5
#define NEDGE  160000
#define NET    6
#define NLAYERS 2

#define LDS_PITCH 36   // bf16 elements per LDS row (32 + 4 pad -> 72B stride)

// ---------------------------------------------------------------------------
// Tiled GEMM: C[M,Ncols] = act(A[M,K] @ B[K,Ncols] + bias)
// Block = 256 threads = 8 waves arranged as 2 M-strips x 4 col-groups:
// block tile 32 rows x 256 cols, grid = (M/32, Ncols/256).
// Per 32-wide K step the block stages A[32][32] as bf16 in LDS (shared by the
// 4 column-groups, cutting A global traffic 4x), then each wave issues
// 4x v_wmma_f32_16x16x32_bf16 on its 16x64 tile.
// B is pre-transposed bf16 Bt[Ncols][K] so B fragments are contiguous 32B.
// A_BF16: 0 = A is f32 (convert in-reg), 1 = A is bf16.
// MODE:   0 = store bf16 with ReLU, 1 = accumulate f32 (out += y).
// Requires M % 32 == 0, K % 32 == 0, Ncols % 256 == 0.
// ---------------------------------------------------------------------------
template<int A_BF16, int MODE>
__global__ __launch_bounds__(256)
void gemm_bf16_wmma(const void* __restrict__ Aptr,
                    const bf16_t* __restrict__ Bt,
                    const float* __restrict__ bias,
                    void* __restrict__ Cptr,
                    int M, int Ncols, int K)
{
    __shared__ bf16_t Alds[32 * LDS_PITCH];

    const int tid  = threadIdx.x;
    const int lane = tid & 31;
    const int wave = tid >> 5;

    const int row0 = blockIdx.x * 32;                    // block row base
    const int m    = (wave & 1) * 16;                    // wave M-strip in block
    const int col0 = blockIdx.y * 256 + (wave >> 1) * 64;// wave col base
    const int r    = lane & 15;
    const int hi   = lane >> 4;

    // cooperative stage coords: each thread loads 4 elements of A[32][32]
    const int lrow = tid >> 3;          // 0..31
    const int kq   = (tid & 7) * 4;     // 0,4,...,28

    v8f acc[4];
    v8f zero = {};
#pragma unroll
    for (int nt = 0; nt < 4; ++nt) acc[nt] = zero;

    for (int k0 = 0; k0 < K; k0 += 32) {
        __syncthreads();                 // protect previous iteration's reads
        // ---- stage A[row0+0..31][k0..k0+31] -> LDS bf16 ----
        if (A_BF16) {
            const bf16_t* A = (const bf16_t*)Aptr + (size_t)(row0 + lrow) * K + k0 + kq;
#pragma unroll
            for (int j = 0; j < 4; ++j) Alds[lrow * LDS_PITCH + kq + j] = A[j];
        } else {
            const float* A = (const float*)Aptr + (size_t)(row0 + lrow) * K + k0 + kq;
            if (k0 + 32 < K) __builtin_prefetch(A + 32, 0, 1);
#pragma unroll
            for (int j = 0; j < 4; ++j) Alds[lrow * LDS_PITCH + kq + j] = (bf16_t)A[j];
        }
        __syncthreads();

        // ---- A fragment from LDS (contiguous k order, same as B):
        // lane holds row lane&15, K = (lane>>4)*16 + [0..15]
        v16bf a;
        {
            const bf16_t* Ar = &Alds[(m + r) * LDS_PITCH + hi * 16];
#pragma unroll
            for (int i = 0; i < 16; ++i) { a[i] = Ar[i]; }
        }

        // ---- B fragments: lane holds column col0+nt*16+(lane&15),
        // K = k0 + (lane>>4)*16 + [0..15] -> contiguous in Bt[col][K]
#pragma unroll
        for (int nt = 0; nt < 4; ++nt) {
            const bf16_t* bp = Bt + (size_t)(col0 + nt * 16 + r) * K + (k0 + hi * 16);
            v16bf b;
#pragma unroll
            for (int j = 0; j < 16; ++j) b[j] = bp[j];
            acc[nt] = __builtin_amdgcn_wmma_f32_16x16x32_bf16(
                false, a, false, b, (short)0, acc[nt], false, false);
        }
    }

    // ---- store: element e of lane l -> C[row0+m + (l>>4)*8 + e][col0 + (l&15)]
#pragma unroll
    for (int nt = 0; nt < 4; ++nt) {
        const int col = col0 + nt * 16 + r;
        const float bv = bias ? bias[col] : 0.0f;
#pragma unroll
        for (int e = 0; e < 8; ++e) {
            const int rr = row0 + m + hi * 8 + e;
            float v = acc[nt][e] + bv;
            if (MODE == 0) {
                ((bf16_t*)Cptr)[(size_t)rr * Ncols + col] = (bf16_t)fmaxf(v, 0.0f);
            } else {
                float* C = (float*)Cptr;
                C[(size_t)rr * Ncols + col] += v;
            }
        }
    }
}

// ---- weight convert + transpose: W[nmat][K][Nc] f32 -> Wt[nmat][Nc][K] bf16
__global__ __launch_bounds__(256)
void convert_transpose(const float* __restrict__ W, bf16_t* __restrict__ Wt,
                       int K, int Nc, int total)
{
    int idx = blockIdx.x * blockDim.x + threadIdx.x;
    if (idx >= total) return;
    int kn = K * Nc;
    int m  = idx / kn;
    int rm = idx - m * kn;
    int k  = rm / Nc;
    int n  = rm - k * Nc;
    Wt[(size_t)m * kn + (size_t)n * K + k] = (bf16_t)W[idx];
}

__global__ __launch_bounds__(256)
void zero_f32(float* __restrict__ p, int n)
{
    int idx = blockIdx.x * blockDim.x + threadIdx.x;
    if (idx < n) p[idx] = 0.0f;
}

// agg = (float)h_dst   (GIN self term, eps=0)
__global__ __launch_bounds__(256)
void init_agg(const bf16_t* __restrict__ hd, float* __restrict__ agg, int n)
{
    int idx = blockIdx.x * blockDim.x + threadIdx.x;
    if (idx < n) agg[idx] = (float)hd[idx];
}

__device__ inline void atomicAddF32(float* p, float v)
{
    __hip_atomic_fetch_add(p, v, __ATOMIC_RELAXED, __HIP_MEMORY_SCOPE_AGENT);
}

// segment-sum: agg[dst[e]] += h_src[src[e]]  (8 features per thread)
__global__ __launch_bounds__(256)
void scatter_edges(const bf16_t* __restrict__ hs,
                   const int* __restrict__ src, const int* __restrict__ dst,
                   float* __restrict__ agg)
{
    const int per = HDIM / 8;                   // 64 threads per edge
    int idx = blockIdx.x * blockDim.x + threadIdx.x;
    if (idx >= NEDGE * per) return;
    int e = idx / per;
    int q = (idx - e * per) * 8;
    int s = src[e];
    int d = dst[e];
    const bf16_t* hp = hs + (size_t)s * HDIM + q;
    float* ap = agg + (size_t)d * HDIM + q;
#pragma unroll
    for (int j = 0; j < 8; ++j) atomicAddF32(ap + j, (float)hp[j]);
}

// h = bf16(relu(out)) over all types
__global__ __launch_bounds__(256)
void relu_to_bf16(const float* __restrict__ o, bf16_t* __restrict__ h, int n)
{
    int idx = blockIdx.x * blockDim.x + threadIdx.x;
    if (idx < n) h[idx] = (bf16_t)fmaxf(o[idx], 0.0f);
}

// y[i] = h0[i] . W_out + b_out  (one wave per node, shuffle reduction)
__global__ __launch_bounds__(256)
void head_kernel(const bf16_t* __restrict__ h0, const float* __restrict__ W_out,
                 const float* __restrict__ b_out, float* __restrict__ y, int n)
{
    int lane = threadIdx.x & 31;
    int wave = threadIdx.x >> 5;
    int node = blockIdx.x * 8 + wave;
    if (node >= n) return;
    const bf16_t* hp = h0 + (size_t)node * HDIM;
    float s = 0.0f;
    for (int f = lane; f < HDIM; f += 32) s += (float)hp[f] * W_out[f];
#pragma unroll
    for (int off = 16; off > 0; off >>= 1) s += __shfl_down(s, off, 32);
    if (lane == 0) y[node] = s + b_out[0];
}

// ---------------------------------------------------------------------------
extern "C" void kernel_launch(void* const* d_in, const int* in_sizes, int n_in,
                              void* d_out, int out_size, void* d_ws, size_t ws_size,
                              hipStream_t stream)
{
    const float* x_all = (const float*)d_in[0];   // [5,N,64]
    const float* W_in  = (const float*)d_in[1];   // [5,64,512]
    const float* b_in  = (const float*)d_in[2];   // [5,512]
    const float* W1    = (const float*)d_in[3];   // [2,6,512,512]
    const float* b1    = (const float*)d_in[4];   // [2,6,512]
    const float* W2    = (const float*)d_in[5];   // [2,6,512,512]
    const float* b2    = (const float*)d_in[6];   // [2,6,512]
    const float* W_out = (const float*)d_in[7];   // [512,1]
    const float* b_out = (const float*)d_in[8];   // [1]
    const int*   ei    = (const int*)d_in[9];     // [6,2,160000]
    float* yout = (float*)d_out;

    const size_t NH = (size_t)NNODES * HDIM;      // 10.24M

    // workspace carve
    char* ws = (char*)d_ws;
    bf16_t* hbf  = (bf16_t*)ws;  ws += (size_t)NTYPES * NH * sizeof(bf16_t);
    float*  outf = (float*)ws;   ws += (size_t)NTYPES * NH * sizeof(float);
    float*  aggf = (float*)ws;   ws += NH * sizeof(float);
    bf16_t* y1bf = (bf16_t*)ws;  ws += NH * sizeof(bf16_t);
    bf16_t* WinT = (bf16_t*)ws;  ws += (size_t)NTYPES * FIN * HDIM * sizeof(bf16_t);
    bf16_t* W1T  = (bf16_t*)ws;  ws += (size_t)NLAYERS * NET * HDIM * HDIM * sizeof(bf16_t);
    bf16_t* W2T  = (bf16_t*)ws;  ws += (size_t)NLAYERS * NET * HDIM * HDIM * sizeof(bf16_t);

    static const int ST[NET] = {4, 1, 0, 2, 3, 1};
    static const int DT[NET] = {1, 0, 2, 3, 0, 4};

    // 1) weight convert + transpose to bf16 [Ncols][K]
    {
        int tot = NTYPES * FIN * HDIM;
        convert_transpose<<<(tot + 255) / 256, 256, 0, stream>>>(W_in, WinT, FIN, HDIM, tot);
        tot = NLAYERS * NET * HDIM * HDIM;
        convert_transpose<<<(tot + 255) / 256, 256, 0, stream>>>(W1, W1T, HDIM, HDIM, tot);
        convert_transpose<<<(tot + 255) / 256, 256, 0, stream>>>(W2, W2T, HDIM, HDIM, tot);
    }

    const dim3 ggrid(NNODES / 32, HDIM / 256);    // 625 x 2 blocks

    // 2) input projection per node type: h = relu(x @ W_in + b_in)
    for (int t = 0; t < NTYPES; ++t) {
        gemm_bf16_wmma<0, 0><<<ggrid, 256, 0, stream>>>(
            x_all + (size_t)t * NNODES * FIN,
            WinT + (size_t)t * FIN * HDIM,
            b_in + (size_t)t * HDIM,
            hbf + (size_t)t * NH,
            NNODES, HDIM, FIN);
    }

    // 3) GIN layers
    const int nNH  = (int)NH;
    const int nAll = NTYPES * (int)NH;
    for (int l = 0; l < NLAYERS; ++l) {
        zero_f32<<<(nAll + 255) / 256, 256, 0, stream>>>(outf, nAll);
        for (int t = 0; t < NET; ++t) {
            const int s = ST[t], d = DT[t];
            const size_t wi = (size_t)(l * NET + t);
            // agg = h[d] + segment_sum(h[s][src] -> dst)
            init_agg<<<(nNH + 255) / 256, 256, 0, stream>>>(hbf + (size_t)d * NH, aggf, nNH);
            {
                int tot = NEDGE * (HDIM / 8);
                scatter_edges<<<(tot + 255) / 256, 256, 0, stream>>>(
                    hbf + (size_t)s * NH,
                    ei + ((size_t)t * 2 + 0) * NEDGE,
                    ei + ((size_t)t * 2 + 1) * NEDGE,
                    aggf);
            }
            // y1 = relu(agg @ W1 + b1)    (store bf16 for next GEMM)
            gemm_bf16_wmma<0, 0><<<ggrid, 256, 0, stream>>>(
                aggf, W1T + wi * HDIM * HDIM, b1 + wi * HDIM, y1bf,
                NNODES, HDIM, HDIM);
            // out[d] += y1 @ W2 + b2
            gemm_bf16_wmma<1, 1><<<ggrid, 256, 0, stream>>>(
                y1bf, W2T + wi * HDIM * HDIM, b2 + wi * HDIM, outf + (size_t)d * NH,
                NNODES, HDIM, HDIM);
        }
        relu_to_bf16<<<(nAll + 255) / 256, 256, 0, stream>>>(outf, hbf, nAll);
    }

    // 4) output head on 'product' nodes (type 0)
    head_kernel<<<(NNODES + 7) / 8, 256, 0, stream>>>(hbf, W_out, b_out, yout, NNODES);
}
